// SpeechAttDecVol7_58884001628649
// MI455X (gfx1250) — matrix operand order repaired and checked by reading, not compiled, not run on hardware
//
#include <hip/hip_runtime.h>
#include <hip/hip_bf16.h>
#include <math.h>

// CDNA5 / gfx1250: wave32, WMMA bf16 16x16x32.
typedef __bf16 v16bf __attribute__((ext_vector_type(16)));
typedef float  v8f   __attribute__((ext_vector_type(8)));

#define L_SEQ  39
#define NBATCH 8192
#define HDIM   256
#define NCHAR  27
#define OUTQ   10
#define BSTRIDE 258   // padded bf16 LDS row stride (dword-aligned, odd dword count -> conflict-free)

// ---- WMMA fragment loaders (ISA 7.12.2 16-bit layouts, wave32) ----
// A (16xK tile, M = lane&15): VGPR v covers K = (v>>2)*16 + half*8 + (v&3)*2 (+kbase)
__device__ __forceinline__ v16bf load_a_frag(const __bf16* p, int stride, int m, int half, int kbase) {
  v16bf a;
#pragma unroll
  for (int v = 0; v < 8; ++v) {
    int k = kbase + ((v >> 2) << 4) + (half << 3) + ((v & 3) << 1);
    a[2 * v]     = p[m * stride + k];
    a[2 * v + 1] = p[m * stride + k + 1];
  }
  return a;
}
// B (Kx16 tile, N = lane&15): VGPR v covers K = (v>>2)*16 + half*8 + (v&3)*2 (+kbase); B[k][n] = W[row=n][k]
__device__ __forceinline__ v16bf load_b_frag(const __bf16* W, int ld, int row, int half, int kbase) {
  v16bf b;
#pragma unroll
  for (int v = 0; v < 8; ++v) {
    int k = kbase + ((v >> 2) << 4) + (half << 3) + ((v & 3) << 1);
    b[2 * v]     = W[row * ld + k];
    b[2 * v + 1] = W[row * ld + k + 1];
  }
  return b;
}
__device__ __forceinline__ v8f wmma_bf16(v16bf a, v16bf b, v8f c) {
  return __builtin_amdgcn_wmma_f32_16x16x32_bf16(false, a, false, b, (short)0, c, false, false);
}

// ---- weight f32 -> bf16 ----
__global__ void f32_to_bf16_kernel(const float* __restrict__ src, __bf16* __restrict__ dst, int n) {
  int i = blockIdx.x * 256 + threadIdx.x;
  if (i < n) dst[i] = (__bf16)src[i];
}

// ---- 39-step GRU recurrence; one workgroup owns 16 batch rows for all steps ----
__global__ __launch_bounds__(128)
void gru_recurrence_kernel(const float* __restrict__ enc, const float* __restrict__ hid0,
                           const float* __restrict__ b_cat, const float* __restrict__ b_ih,
                           const float* __restrict__ b_hh,
                           const __bf16* __restrict__ Wcat, const __bf16* __restrict__ Wih,
                           const __bf16* __restrict__ Whh, __bf16* __restrict__ outs)
{
  __shared__ float  hidf[16 * HDIM];           // f32 hidden (for exact blend)
  __shared__ __bf16 hidb[2][16 * BSTRIDE];     // bf16 hidden, double-buffered
  __shared__ __bf16 xb[16 * BSTRIDE];          // bf16 x (also temp for ctx)
  __shared__ float  xc[16 * HDIM];             // per-row constant ctx@Wcc^T + b_cat

  const int nbase = blockIdx.x * 16;
  const int tid  = threadIdx.x;
  const int wave = tid >> 5;
  const int lane = tid & 31;
  const int lm = lane & 15;   // A: row m / B: col n / D: col n
  const int lh = lane >> 4;   // lane half

  // init hidden; ctx = mean_l enc[l, n, :]   (attention is uniform 1/L)
  for (int i = tid; i < 16 * HDIM; i += 128) {
    int m = i >> 8, h = i & 255;
    float v = hid0[(size_t)(nbase + m) * HDIM + h];
    hidf[m * HDIM + h] = v;
    hidb[0][m * BSTRIDE + h] = (__bf16)v;
    float s = 0.f;
    for (int l = 0; l < L_SEQ; ++l)
      s += enc[((size_t)l * NBATCH + nbase + m) * HDIM + h];
    xb[m * BSTRIDE + h] = (__bf16)(s * (1.0f / (float)L_SEQ));
  }
  __syncthreads();

  // xc = ctx @ W_cat[:, H:2H]^T + b_cat   (constant over timesteps)
  for (int t = 0; t < 4; ++t) {
    const int cb = (wave * 4 + t) * 16;
    v8f acc = {};
    for (int kk = 0; kk < HDIM; kk += 32) {
      v16bf a = load_a_frag(xb, BSTRIDE, lm, lh, kk);
      v16bf b = load_b_frag(Wcat, 2 * HDIM, cb + lm, lh, kk + HDIM);
      acc = wmma_bf16(a, b, acc);
    }
    float bc = b_cat[cb + lm];
#pragma unroll
    for (int r = 0; r < 8; ++r)
      xc[(r + 8 * lh) * HDIM + cb + lm] = acc[r] + bc;
  }
  __syncthreads();

  int cur = 0;
  for (int l = 0; l < L_SEQ; ++l) {
    // Phase A: x = hid @ W_cat[:, 0:H]^T + xc  -> xb (bf16)
    for (int t = 0; t < 4; ++t) {
      const int cb = (wave * 4 + t) * 16;
      v8f acc = {};
      for (int kk = 0; kk < HDIM; kk += 32) {
        v16bf a = load_a_frag(hidb[cur], BSTRIDE, lm, lh, kk);
        v16bf b = load_b_frag(Wcat, 2 * HDIM, cb + lm, lh, kk);
        acc = wmma_bf16(a, b, acc);
      }
#pragma unroll
      for (int r = 0; r < 8; ++r) {
        int m = r + 8 * lh;
        xb[m * BSTRIDE + cb + lm] = (__bf16)(acc[r] + xc[m * HDIM + cb + lm]);
      }
    }
    __syncthreads();

    // Phase B: six gate GEMMs (r,z,n from x and from hid), combine, write new hid + outs
    for (int t = 0; t < 4; ++t) {
      const int cb = (wave * 4 + t) * 16;
      v8f ir = {}, iz = {}, in2 = {}, hr = {}, hz = {}, hn = {};
      for (int kk = 0; kk < HDIM; kk += 32) {
        v16bf ax = load_a_frag(xb, BSTRIDE, lm, lh, kk);
        v16bf ah = load_a_frag(hidb[cur], BSTRIDE, lm, lh, kk);
        ir  = wmma_bf16(ax, load_b_frag(Wih, HDIM,            cb + lm, lh, kk), ir);
        iz  = wmma_bf16(ax, load_b_frag(Wih, HDIM,     HDIM + cb + lm, lh, kk), iz);
        in2 = wmma_bf16(ax, load_b_frag(Wih, HDIM, 2 * HDIM + cb + lm, lh, kk), in2);
        hr  = wmma_bf16(ah, load_b_frag(Whh, HDIM,            cb + lm, lh, kk), hr);
        hz  = wmma_bf16(ah, load_b_frag(Whh, HDIM,     HDIM + cb + lm, lh, kk), hz);
        hn  = wmma_bf16(ah, load_b_frag(Whh, HDIM, 2 * HDIM + cb + lm, lh, kk), hn);
      }
      const float bir = b_ih[cb + lm],            bhr = b_hh[cb + lm];
      const float biz = b_ih[HDIM + cb + lm],     bhz = b_hh[HDIM + cb + lm];
      const float bin = b_ih[2 * HDIM + cb + lm], bhn = b_hh[2 * HDIM + cb + lm];
#pragma unroll
      for (int r = 0; r < 8; ++r) {
        int m = r + 8 * lh;
        float rg = 1.f / (1.f + __expf(-(ir[r] + bir + hr[r] + bhr)));
        float zg = 1.f / (1.f + __expf(-(iz[r] + biz + hz[r] + bhz)));
        float ng = tanhf(in2[r] + bin + rg * (hn[r] + bhn));
        float ho = hidf[m * HDIM + cb + lm];      // this wave alone owns this element
        float hNew = (1.f - zg) * ng + zg * ho;
        hidf[m * HDIM + cb + lm] = hNew;
        hidb[cur ^ 1][m * BSTRIDE + cb + lm] = (__bf16)hNew;
        outs[((size_t)l * NBATCH + nbase + m) * HDIM + cb + lm] = (__bf16)hNew;
      }
    }
    cur ^= 1;
    __syncthreads();
  }
}

// ---- tail: S[n,o,:] = sum_l Wq[o,l]*outs[l,n,:]; logits = S@W_out^T + bias; clamp; log_softmax(27) ----
__global__ __launch_bounds__(256)
void finalize_kernel(const __bf16* __restrict__ outs, const float* __restrict__ W_out,
                     const float* __restrict__ b_out, const float* __restrict__ W_q,
                     const float* __restrict__ b_q, float* __restrict__ out)
{
  __shared__ float S[4 * OUTQ * HDIM];    // 40KB
  __shared__ float T[4 * OUTQ * NCHAR];
  __shared__ float wq[OUTQ * L_SEQ];
  __shared__ float swq[OUTQ];

  const int nbase = blockIdx.x * 4;
  const int tid = threadIdx.x;

  for (int i = tid; i < OUTQ * L_SEQ; i += 256) wq[i] = W_q[i];
  __syncthreads();
  if (tid < OUTQ) {
    float s = 0.f;
    for (int l = 0; l < L_SEQ; ++l) s += wq[tid * L_SEQ + l];
    swq[tid] = s;
  }

  const int m  = tid >> 6;        // 0..3
  const int hb = (tid & 63) * 4;  // 4 h per thread
  float acc[OUTQ][4];
#pragma unroll
  for (int o = 0; o < OUTQ; ++o)
#pragma unroll
    for (int j = 0; j < 4; ++j) acc[o][j] = 0.f;

  for (int l = 0; l < L_SEQ; ++l) {
    float v[4];
#pragma unroll
    for (int j = 0; j < 4; ++j)
      v[j] = (float)outs[((size_t)l * NBATCH + nbase + m) * HDIM + hb + j];
#pragma unroll
    for (int o = 0; o < OUTQ; ++o) {
      float w = wq[o * L_SEQ + l];
#pragma unroll
      for (int j = 0; j < 4; ++j) acc[o][j] += w * v[j];
    }
  }
#pragma unroll
  for (int o = 0; o < OUTQ; ++o)
#pragma unroll
    for (int j = 0; j < 4; ++j)
      S[(m * OUTQ + o) * HDIM + hb + j] = acc[o][j];
  __syncthreads();

  for (int i = tid; i < 4 * OUTQ * NCHAR; i += 256) {
    int c = i % NCHAR;
    int mo = i / NCHAR;
    int o = mo % OUTQ;
    float s = 0.f;
    for (int h = 0; h < HDIM; ++h)
      s += S[mo * HDIM + h] * W_out[c * HDIM + h];
    T[i] = s + b_out[c] * swq[o] + b_q[o];
  }
  __syncthreads();

  for (int row = tid; row < 4 * OUTQ; row += 256) {
    float mx = -1e30f;
    for (int c = 0; c < NCHAR; ++c)
      mx = fmaxf(mx, fmaxf(T[row * NCHAR + c], 1e-4f));
    float se = 0.f;
    for (int c = 0; c < NCHAR; ++c)
      se += __expf(fmaxf(T[row * NCHAR + c], 1e-4f) - mx);
    float lse = __logf(se) + mx;
    int mm = row / OUTQ, o = row % OUTQ;
    for (int c = 0; c < NCHAR; ++c)
      out[((size_t)(nbase + mm) * OUTQ + o) * NCHAR + c] =
          fmaxf(T[row * NCHAR + c], 1e-4f) - lse;
  }
}

extern "C" void kernel_launch(void* const* d_in, const int* in_sizes, int n_in,
                              void* d_out, int out_size, void* d_ws, size_t ws_size,
                              hipStream_t stream) {
  const float* enc   = (const float*)d_in[0];
  const float* hid0  = (const float*)d_in[1];
  const float* W_cat = (const float*)d_in[2];
  const float* b_cat = (const float*)d_in[3];
  const float* W_ih  = (const float*)d_in[4];
  const float* W_hh  = (const float*)d_in[5];
  const float* b_ih  = (const float*)d_in[6];
  const float* b_hh  = (const float*)d_in[7];
  const float* W_out = (const float*)d_in[8];
  const float* b_out = (const float*)d_in[9];
  const float* W_q   = (const float*)d_in[10];
  const float* b_q   = (const float*)d_in[11];
  float* out = (float*)d_out;

  // workspace layout: bf16 weights (1 MB) + bf16 outs (39*8192*256*2 ~ 156 MB)
  char* ws = (char*)d_ws;
  const size_t szWcat = (size_t)256 * 512 * 2;
  const size_t szWih  = (size_t)768 * 256 * 2;
  __bf16* Wcat_bf = (__bf16*)(ws);
  __bf16* Wih_bf  = (__bf16*)(ws + szWcat);
  __bf16* Whh_bf  = (__bf16*)(ws + szWcat + szWih);
  __bf16* outs_bf = (__bf16*)(ws + szWcat + 2 * szWih);

  f32_to_bf16_kernel<<<(256 * 512 + 255) / 256, 256, 0, stream>>>(W_cat, Wcat_bf, 256 * 512);
  f32_to_bf16_kernel<<<(768 * 256 + 255) / 256, 256, 0, stream>>>(W_ih, Wih_bf, 768 * 256);
  f32_to_bf16_kernel<<<(768 * 256 + 255) / 256, 256, 0, stream>>>(W_hh, Whh_bf, 768 * 256);

  gru_recurrence_kernel<<<NBATCH / 16, 128, 0, stream>>>(enc, hid0, b_cat, b_ih, b_hh,
                                                         Wcat_bf, Wih_bf, Whh_bf, outs_bf);
  finalize_kernel<<<NBATCH / 4, 256, 0, stream>>>(outs_bf, W_out, b_out, W_q, b_q, out);
}
